// GPT_66451734004244
// MI455X (gfx1250) — matrix operand order repaired and checked by reading, not compiled, not run on hardware
//
#include <hip/hip_runtime.h>
#include <hip/hip_bf16.h>
#include <math.h>

// ---------------- model constants ----------------
#define VOCAB 32000
#define TBLK  1024
#define NLAYER 4
#define NHEAD 12
#define EMB   768
#define HDIM  64          // EMB / NHEAD
#define FFDIM 3072        // 4*EMB
#define E3    2304        // 3*EMB
#define BATCH 2
#define MROWS (BATCH*TBLK)   // 2048

// wave tile: WM x WN grid of 16x16 WMMA tiles => 32 x 64 output per wave
#define WM 2
#define WN 4
// 8 waves per block stacked along M => block covers 256 x 64
#define BLK_M (WM*16*8)   // 256
#define BLK_N (WN*16)     // 64

typedef __attribute__((ext_vector_type(16))) _Float16 v16h;
typedef __attribute__((ext_vector_type(8)))  float    v8f;
typedef __attribute__((ext_vector_type(8)))  unsigned int v8u;

// ---------------- fp32 -> fp16 convert ----------------
__global__ void cvt_f16_kernel(const float* __restrict__ src,
                               _Float16* __restrict__ dst, long n) {
    long i = (long)blockIdx.x * blockDim.x + threadIdx.x;
    if (i < n) dst[i] = (_Float16)src[i];
}

// ---------------- embedding: x = w_emb[idx] + w_pos ----------------
__global__ void embed_kernel(const int* __restrict__ idx,
                             const float* __restrict__ w_emb,
                             const float* __restrict__ w_pos,
                             float* __restrict__ x) {
    long i = (long)blockIdx.x * blockDim.x + threadIdx.x;
    if (i >= (long)MROWS * EMB) return;
    int row = (int)(i / EMB);
    int col = (int)(i % EMB);
    int tok = idx[row];
    int t   = row % TBLK;
    x[i] = w_emb[(long)tok * EMB + col] + w_pos[(long)t * EMB + col];
}

// ---------------- layernorm (fp32 in -> f16 out) ----------------
__global__ void ln_kernel(const float* __restrict__ in,
                          const float* __restrict__ g,
                          const float* __restrict__ b,
                          _Float16* __restrict__ out, int N) {
    int row = blockIdx.x, tid = threadIdx.x;
    __shared__ float red[256];
    const float* x = in + (long)row * N;
    float s = 0.f;
    for (int i = tid; i < N; i += 256) s += x[i];
    red[tid] = s; __syncthreads();
    for (int o = 128; o; o >>= 1) { if (tid < o) red[tid] += red[tid + o]; __syncthreads(); }
    float mu = red[0] / N; __syncthreads();
    float v = 0.f;
    for (int i = tid; i < N; i += 256) { float d = x[i] - mu; v += d * d; }
    red[tid] = v; __syncthreads();
    for (int o = 128; o; o >>= 1) { if (tid < o) red[tid] += red[tid + o]; __syncthreads(); }
    float rstd = rsqrtf(red[0] / N + 1e-5f);
    for (int i = tid; i < N; i += 256)
        out[(long)row * N + i] = (_Float16)((x[i] - mu) * rstd * g[i] + b[i]);
}

// ---------------- fragment loader: 8 f16-pairs, two b128 chunks ----------------
__device__ __forceinline__ v16h load_frag_a(const unsigned int* p) {
    uint4 a0 = *(const uint4*)(p);
    uint4 a1 = *(const uint4*)(p + 8);     // A: K+16..23 group lives 8 pairs on
    v8u u;
    u[0]=a0.x; u[1]=a0.y; u[2]=a0.z; u[3]=a0.w;
    u[4]=a1.x; u[5]=a1.y; u[6]=a1.z; u[7]=a1.w;
    return __builtin_bit_cast(v16h, u);
}
__device__ __forceinline__ v16h load_frag_b(const unsigned int* p) {
    uint4 b0 = *(const uint4*)(p);
    uint4 b1 = *(const uint4*)(p + 8);     // B: K+16..23 group lives 8 pairs on
    v8u u;
    u[0]=b0.x; u[1]=b0.y; u[2]=b0.z; u[3]=b0.w;
    u[4]=b1.x; u[5]=b1.y; u[6]=b1.z; u[7]=b1.w;
    return __builtin_bit_cast(v16h, u);
}

// ---------------- WMMA GEMM: C[M,N] = A[M,K] * W[N,K]^T ----------------
// Register-tiled: each wave computes 32x64 (2x4 WMMA tiles); 8 waves/block
// stacked along M share identical B fragments (weight stream L0-resident).
__global__ void gemm_wmma_kernel(const _Float16* __restrict__ A,
                                 const _Float16* __restrict__ W,
                                 float* __restrict__ C,
                                 int M, int N, int K) {
    const int lane = threadIdx.x & 31;
    const int wv   = threadIdx.x >> 5;            // 0..7
    const int blocks_m = M / BLK_M;
    const int bm = blockIdx.x % blocks_m;
    const int bn = blockIdx.x / blocks_m;
    const int mbase = (bm * 8 + wv) * (WM * 16);
    const int nbase = bn * BLK_N;
    if (mbase >= M || nbase >= N) return;         // wave-uniform

    const unsigned int* Au = (const unsigned int*)A;   // f16 pairs
    const unsigned int* Wu = (const unsigned int*)W;
    const int K2 = K >> 1;

    const int aSel = (lane >> 4) * 8;             // A lane-half K offset
    const int bSel = (lane >> 4) * 8;             // B lane-half K offset
    const int l15  = lane & 15;

    const unsigned int* aBase[WM];
    const unsigned int* bBase[WN];
#pragma unroll
    for (int mt = 0; mt < WM; ++mt)
        aBase[mt] = Au + (long)(mbase + mt * 16 + l15) * K2 + (aSel >> 1);
#pragma unroll
    for (int nt = 0; nt < WN; ++nt)
        bBase[nt] = Wu + (long)(nbase + nt * 16 + l15) * K2 + (bSel >> 1);

    v8f acc[WM][WN] = {};
    for (int k0 = 0; k0 < K; k0 += 32) {
        const int kp = k0 >> 1;                   // pair offset
        v16h aF[WM], bF[WN];
#pragma unroll
        for (int mt = 0; mt < WM; ++mt) {
            __builtin_prefetch(aBase[mt] + kp + 16, 0, 0);
            aF[mt] = load_frag_a(aBase[mt] + kp);
        }
#pragma unroll
        for (int nt = 0; nt < WN; ++nt) {
            __builtin_prefetch(bBase[nt] + kp + 16, 0, 0);
            bF[nt] = load_frag_b(bBase[nt] + kp);
        }
#pragma unroll
        for (int mt = 0; mt < WM; ++mt)
#pragma unroll
            for (int nt = 0; nt < WN; ++nt)
                acc[mt][nt] = __builtin_amdgcn_wmma_f32_16x16x32_f16(
                    false, aF[mt], false, bF[nt],
                    (short)0, acc[mt][nt], false, false);
    }

    const int rhalf = (lane >> 4) * 8;            // C: M split across lane halves
#pragma unroll
    for (int mt = 0; mt < WM; ++mt) {
        const int crow0 = mbase + mt * 16 + rhalf;
#pragma unroll
        for (int nt = 0; nt < WN; ++nt) {
            const int ccol = nbase + nt * 16 + l15;
#pragma unroll
            for (int r = 0; r < 8; ++r)
                C[(long)(crow0 + r) * N + ccol] = acc[mt][nt][r];
        }
    }
}

// ---------------- causal attention, online softmax, wave per (b,h,q) ----------------
__global__ void attn_kernel(const float* __restrict__ qkv,   // [MROWS, 3E]
                            _Float16* __restrict__ y) {      // [MROWS, E]
    int gw   = (blockIdx.x * blockDim.x + threadIdx.x) >> 5;
    int lane = threadIdx.x & 31;
    if (gw >= BATCH * NHEAD * TBLK) return;
    int q = gw % TBLK;
    int h = (gw / TBLK) % NHEAD;
    int b = gw / (TBLK * NHEAD);
    const float scale = 0.125f;                   // 1/sqrt(64)
    const float* qrow = qkv + (long)(b * TBLK + q) * E3 + h * HDIM;
    float q0 = qrow[lane] * scale, q1 = qrow[lane + 32] * scale;
    float m = -INFINITY, s = 0.f, a0 = 0.f, a1 = 0.f;
    for (int k = 0; k <= q; ++k) {
        const float* krow = qkv + (long)(b * TBLK + k) * E3 + EMB + h * HDIM;
        float p = q0 * krow[lane] + q1 * krow[lane + 32];
        for (int o = 16; o; o >>= 1) p += __shfl_xor(p, o, 32);
        float nm = fmaxf(m, p);
        float f  = expf(m - nm);
        float e  = expf(p - nm);
        const float* vrow = qkv + (long)(b * TBLK + k) * E3 + 2 * EMB + h * HDIM;
        s  = s  * f + e;
        a0 = a0 * f + e * vrow[lane];
        a1 = a1 * f + e * vrow[lane + 32];
        m = nm;
    }
    float inv = 1.0f / s;
    _Float16* yo = y + (long)(b * TBLK + q) * EMB + h * HDIM;
    yo[lane]      = (_Float16)(a0 * inv);
    yo[lane + 32] = (_Float16)(a1 * inv);
}

// ---------------- bias + exact GELU -> f16 ----------------
__global__ void bias_gelu_kernel(const float* __restrict__ hin,
                                 const float* __restrict__ bias,
                                 _Float16* __restrict__ hout, long n, int N) {
    long i = (long)blockIdx.x * blockDim.x + threadIdx.x;
    if (i >= n) return;
    float v = hin[i] + bias[i % N];
    hout[i] = (_Float16)(0.5f * v * (1.0f + erff(v * 0.70710678118f)));
}

// ---------------- residual add (+ optional bias) ----------------
__global__ void resadd_kernel(float* __restrict__ x, const float* __restrict__ d,
                              const float* __restrict__ bias, long n, int N) {
    long i = (long)blockIdx.x * blockDim.x + threadIdx.x;
    if (i >= n) return;
    float v = x[i] + d[i];
    if (bias) v += bias[i % N];
    x[i] = v;
}

// ---------------- per-row NLL over logits ----------------
__global__ void loss_row_kernel(const float* __restrict__ logits,
                                const int* __restrict__ tgt,
                                float* __restrict__ nll, int Vn) {
    int row = blockIdx.x, tid = threadIdx.x;
    const float* l = logits + (long)row * Vn;
    __shared__ float red[256];
    float mx = -INFINITY;
    for (int i = tid; i < Vn; i += 256) mx = fmaxf(mx, l[i]);
    red[tid] = mx; __syncthreads();
    for (int o = 128; o; o >>= 1) { if (tid < o) red[tid] = fmaxf(red[tid], red[tid + o]); __syncthreads(); }
    mx = red[0]; __syncthreads();
    float s = 0.f;
    for (int i = tid; i < Vn; i += 256) s += expf(l[i] - mx);
    red[tid] = s; __syncthreads();
    for (int o = 128; o; o >>= 1) { if (tid < o) red[tid] += red[tid + o]; __syncthreads(); }
    if (tid == 0) nll[row] = (logf(red[0]) + mx) - l[tgt[row]];
}

__global__ void loss_mean_kernel(const float* __restrict__ nll, int n,
                                 float* __restrict__ out) {
    __shared__ float red[256];
    int tid = threadIdx.x;
    float s = 0.f;
    for (int i = tid; i < n; i += 256) s += nll[i];
    red[tid] = s; __syncthreads();
    for (int o = 128; o; o >>= 1) { if (tid < o) red[tid] += red[tid + o]; __syncthreads(); }
    if (tid == 0) *out = red[0] / n;
}

// ---------------- host side ----------------
static inline int eltBlocks(long n) { return (int)((n + 255) / 256); }
static inline int gemmBlocks(int M, int N) { return (M / BLK_M) * (N / BLK_N); }

extern "C" void kernel_launch(void* const* d_in, const int* in_sizes, int n_in,
                              void* d_out, int out_size, void* d_ws, size_t ws_size,
                              hipStream_t stream) {
    const int*   idx     = (const int*)  d_in[0];
    const int*   tgt     = (const int*)  d_in[1];
    const float* w_emb   = (const float*)d_in[2];
    const float* w_pos   = (const float*)d_in[3];
    const float* ln1_w   = (const float*)d_in[4];
    const float* ln1_b   = (const float*)d_in[5];
    const float* qkv_w   = (const float*)d_in[6];
    const float* aproj_w = (const float*)d_in[7];
    const float* ln2_w   = (const float*)d_in[8];
    const float* ln2_b   = (const float*)d_in[9];
    const float* fc_w    = (const float*)d_in[10];
    const float* fc_b    = (const float*)d_in[11];
    const float* mproj_w = (const float*)d_in[12];
    const float* mproj_b = (const float*)d_in[13];
    const float* lnf_w   = (const float*)d_in[14];
    const float* lnf_b   = (const float*)d_in[15];

    // workspace carve (256B aligned)
    char* ws = (char*)d_ws;
    size_t off = 0;
    auto carve = [&](size_t bytes) -> void* {
        void* p = ws + off; off += (bytes + 255) & ~(size_t)255; return p;
    };
    _Float16* w_emb16   = (_Float16*)carve((size_t)VOCAB * EMB * 2);
    _Float16* qkv_w16   = (_Float16*)carve((size_t)NLAYER * E3 * EMB * 2);
    _Float16* aproj_w16 = (_Float16*)carve((size_t)NLAYER * EMB * EMB * 2);
    _Float16* fc_w16    = (_Float16*)carve((size_t)NLAYER * FFDIM * EMB * 2);
    _Float16* mproj_w16 = (_Float16*)carve((size_t)NLAYER * EMB * FFDIM * 2);
    float*    x         = (float*)   carve((size_t)MROWS * EMB * 4);
    _Float16* a16       = (_Float16*)carve((size_t)MROWS * FFDIM * 2); // GEMM A staging
    float*    qkv32     = (float*)   carve((size_t)MROWS * E3 * 4);
    float*    tmp32     = (float*)   carve((size_t)MROWS * FFDIM * 4);
    float*    nll       = (float*)   carve((size_t)MROWS * 4);

    float* logits = (float*)d_out;                    // [MROWS, VOCAB]
    float* loss   = logits + (size_t)MROWS * VOCAB;   // scalar

    // --- weight conversion to f16 ---
    long n;
    n = (long)VOCAB * EMB;
    cvt_f16_kernel<<<eltBlocks(n), 256, 0, stream>>>(w_emb, w_emb16, n);
    n = (long)NLAYER * E3 * EMB;
    cvt_f16_kernel<<<eltBlocks(n), 256, 0, stream>>>(qkv_w, qkv_w16, n);
    n = (long)NLAYER * EMB * EMB;
    cvt_f16_kernel<<<eltBlocks(n), 256, 0, stream>>>(aproj_w, aproj_w16, n);
    n = (long)NLAYER * FFDIM * EMB;
    cvt_f16_kernel<<<eltBlocks(n), 256, 0, stream>>>(fc_w, fc_w16, n);
    n = (long)NLAYER * EMB * FFDIM;
    cvt_f16_kernel<<<eltBlocks(n), 256, 0, stream>>>(mproj_w, mproj_w16, n);

    // --- embedding ---
    n = (long)MROWS * EMB;
    embed_kernel<<<eltBlocks(n), 256, 0, stream>>>(idx, w_emb, w_pos, x);

    // --- transformer layers ---
    for (int l = 0; l < NLAYER; ++l) {
        // ln1 -> a16
        ln_kernel<<<MROWS, 256, 0, stream>>>(x, ln1_w + (size_t)l * EMB,
                                             ln1_b + (size_t)l * EMB, a16, EMB);
        // qkv = ln1(x) @ qkv_w^T
        gemm_wmma_kernel<<<gemmBlocks(MROWS, E3), 256, 0, stream>>>(
            a16, qkv_w16 + (size_t)l * E3 * EMB, qkv32, MROWS, E3, EMB);
        // attention -> a16 (f16 y)
        attn_kernel<<<(BATCH * NHEAD * TBLK * 32) / 256, 256, 0, stream>>>(qkv32, a16);
        // attn proj
        gemm_wmma_kernel<<<gemmBlocks(MROWS, EMB), 256, 0, stream>>>(
            a16, aproj_w16 + (size_t)l * EMB * EMB, tmp32, MROWS, EMB, EMB);
        n = (long)MROWS * EMB;
        resadd_kernel<<<eltBlocks(n), 256, 0, stream>>>(x, tmp32, nullptr, n, EMB);
        // ln2 -> a16
        ln_kernel<<<MROWS, 256, 0, stream>>>(x, ln2_w + (size_t)l * EMB,
                                             ln2_b + (size_t)l * EMB, a16, EMB);
        // fc
        gemm_wmma_kernel<<<gemmBlocks(MROWS, FFDIM), 256, 0, stream>>>(
            a16, fc_w16 + (size_t)l * FFDIM * EMB, tmp32, MROWS, FFDIM, EMB);
        n = (long)MROWS * FFDIM;
        bias_gelu_kernel<<<eltBlocks(n), 256, 0, stream>>>(
            tmp32, fc_b + (size_t)l * FFDIM, a16, n, FFDIM);
        // mlp proj
        gemm_wmma_kernel<<<gemmBlocks(MROWS, EMB), 256, 0, stream>>>(
            a16, mproj_w16 + (size_t)l * EMB * FFDIM, tmp32, MROWS, EMB, FFDIM);
        n = (long)MROWS * EMB;
        resadd_kernel<<<eltBlocks(n), 256, 0, stream>>>(
            x, tmp32, mproj_b + (size_t)l * EMB, n, EMB);
    }

    // --- final LN + tied LM head ---
    ln_kernel<<<MROWS, 256, 0, stream>>>(x, lnf_w, lnf_b, a16, EMB);
    gemm_wmma_kernel<<<gemmBlocks(MROWS, VOCAB), 256, 0, stream>>>(
        a16, w_emb16, logits, MROWS, VOCAB, EMB);

    // --- loss ---
    loss_row_kernel<<<MROWS, 256, 0, stream>>>(logits, tgt, nll, VOCAB);
    loss_mean_kernel<<<1, 256, 0, stream>>>(nll, MROWS, loss);
}
